// RelativeDecoderBlock_18580028522813
// MI455X (gfx1250) — matrix operand order repaired and checked by reading, not compiled, not run on hardware
//
#include <hip/hip_runtime.h>
#include <hip/hip_bf16.h>

// ---------------------------------------------------------------------------
// Types & helpers
// ---------------------------------------------------------------------------
typedef __attribute__((ext_vector_type(16))) __bf16 v16bf;
typedef __attribute__((ext_vector_type(8)))  float  v8f;

union FragBF { v16bf v; unsigned int u[8]; };

__device__ __forceinline__ unsigned short f2bf(float f) {
    unsigned int u = __float_as_uint(f);
    unsigned int r = u + 0x7fffu + ((u >> 16) & 1u);   // round-to-nearest-even
    return (unsigned short)(r >> 16);
}
__device__ __forceinline__ float bf2f(unsigned short h) {
    return __uint_as_float(((unsigned int)h) << 16);
}

// Problem constants
#define BN_  8
#define PC_  128
#define FC_  128
#define HH_  64
#define WW_  64
#define HW_  4096
#define LC_  8
#define KK_  512
#define INC_ 256
#define DD_  32768          // LC*H*W
#define MM_  32768          // B*H*W

// ---------------------------------------------------------------------------
// WMMA bf16 GEMM:  Y[M][N] = A[M][K] * B[K][N] (+bias[N])
// Requirements: M%128==0, N%128==0, K%32==0, 16B-aligned pointers.
// Block = 8 waves; wave w computes rows [16w,16w+16) x all 128 cols.
// ---------------------------------------------------------------------------
#define GTM 128
#define GTN 128
#define GTK 32
#define APAD 8              // sA row stride 40 ushorts = 80 B (16B multiple)
#define BPAD 24             // sBt row stride 56 ushorts = 112 B (16B multiple)

__global__ __launch_bounds__(256) void k_gemm(
    const unsigned short* __restrict__ A,
    const unsigned short* __restrict__ Bm,
    const float* __restrict__ bias,
    float* __restrict__ Y,
    int M, int N, int Kd)
{
    __shared__ unsigned short sA [2][GTM][GTK + APAD];
    __shared__ unsigned short sBt[2][GTN][GTK + BPAD];

    const int tid  = threadIdx.x;
    const int lane = tid & 31;
    const int wave = tid >> 5;             // 0..7
    const int m0   = blockIdx.y * GTM;
    const int n0   = blockIdx.x * GTN;
    const int g    = lane >> 4;            // lane half

    v8f acc[8];
#pragma unroll
    for (int i = 0; i < 8; ++i)
#pragma unroll
        for (int j = 0; j < 8; ++j) acc[i][j] = 0.0f;

    const int nK = Kd / GTK;

    auto loadTiles = [&](int s, int k0) {
        { // A tile: 128 rows x 32 cols (coalesced 16B loads, aligned 16B LDS stores)
            int row = tid >> 1;
            int col = (tid & 1) * 16;
            const unsigned short* gp = A + (size_t)(m0 + row) * Kd + k0 + col;
            uint4 v0 = *(const uint4*)(gp);
            uint4 v1 = *(const uint4*)(gp + 8);
            *(uint4*)&sA[s][row][col]     = v0;
            *(uint4*)&sA[s][row][col + 8] = v1;
            __builtin_prefetch(gp + GTK, 0, 1);     // global_prefetch next K tile
        }
        { // B tile: 32 rows x 128 cols, stored transposed sBt[n][k]
            int krow = tid >> 3;
            int nseg = (tid & 7) * 16;
            const unsigned short* gp = Bm + (size_t)(k0 + krow) * N + n0 + nseg;
            uint4 v0 = *(const uint4*)(gp);
            uint4 v1 = *(const uint4*)(gp + 8);
            const unsigned short* p0 = (const unsigned short*)&v0;
            const unsigned short* p1 = (const unsigned short*)&v1;
#pragma unroll
            for (int j = 0; j < 8; ++j) sBt[s][nseg + j][krow]     = p0[j];
#pragma unroll
            for (int j = 0; j < 8; ++j) sBt[s][nseg + 8 + j][krow] = p1[j];
        }
    };

    loadTiles(0, 0);
    __syncthreads();

    const int mrow = wave * 16 + (lane & 15);
    const int ncl  = lane & 15;

    for (int kt = 0; kt < nK; ++kt) {
        int cur = kt & 1;
        if (kt + 1 < nK) loadTiles((kt + 1) & 1, (kt + 1) * GTK);

        // A fragment (16x32 bf16, wave32 layout): two contiguous 16B runs
        FragBF af;
#pragma unroll
        for (int v = 0; v < 8; ++v) {
            int kb = (v < 4) ? (2 * v + 8 * g) : (16 + 2 * (v - 4) + 8 * g);
            af.u[v] = *(const unsigned int*)&sA[cur][mrow][kb];
        }
        // Preload ALL 8 B fragments into distinct registers, THEN run the
        // WMMA chain back-to-back: no WAR hazard NOPs between WMMAs, one
        // batched LDS wait instead of one wait per WMMA.
        FragBF bf8[8];
#pragma unroll
        for (int nt = 0; nt < 8; ++nt) {
            int ncol = nt * 16 + ncl;
#pragma unroll
            for (int v = 0; v < 8; ++v) {
                int kb = (v < 4) ? (2 * v + 8 * g) : (16 + 2 * (v - 4) + 8 * g);  // table order, matches A
                bf8[nt].u[v] = *(const unsigned int*)&sBt[cur][ncol][kb];
            }
        }
#pragma unroll
        for (int nt = 0; nt < 8; ++nt) {
            acc[nt] = __builtin_amdgcn_wmma_f32_16x16x32_bf16(
                false, af.v, false, bf8[nt].v, (short)0, acc[nt], false, false);
        }
        __syncthreads();
    }

    // store D (C/D layout: vgpr r, lane -> n=lane%16, m=r+8*(lane/16))
#pragma unroll
    for (int nt = 0; nt < 8; ++nt) {
        int nn = n0 + nt * 16 + ncl;
        float bv = bias ? bias[nn] : 0.0f;
#pragma unroll
        for (int r = 0; r < 8; ++r) {
            int mm = m0 + wave * 16 + r + 8 * g;
            Y[(size_t)mm * N + nn] = acc[nt][r] + bv;
        }
    }
}

// ---------------------------------------------------------------------------
// BN statistics (per channel, NCHW input)
// ---------------------------------------------------------------------------
__global__ void k_channel_stats(const float* __restrict__ x, int Cin, int HW, int Bn,
                                float* __restrict__ mean, float* __restrict__ rstd)
{
    int c = blockIdx.x;
    float s = 0.f, s2 = 0.f;
    for (int b = 0; b < Bn; ++b) {
        const float* p = x + ((size_t)b * Cin + c) * HW;
        for (int i = threadIdx.x; i < HW; i += 256) { float v = p[i]; s += v; s2 += v * v; }
    }
    __shared__ float r1[256], r2[256];
    r1[threadIdx.x] = s; r2[threadIdx.x] = s2;
    __syncthreads();
    for (int st = 128; st > 0; st >>= 1) {
        if (threadIdx.x < st) { r1[threadIdx.x] += r1[threadIdx.x + st]; r2[threadIdx.x] += r2[threadIdx.x + st]; }
        __syncthreads();
    }
    if (threadIdx.x == 0) {
        float cnt = (float)(Bn * HW);
        float m = r1[0] / cnt;
        float var = r2[0] / cnt - m * m;
        mean[c] = m; rstd[c] = rsqrtf(var + 1e-5f);
    }
}

// NCHW float -> NHWC bf16, BN-affine (bn0) plus raw copy; LDS 32x32 transpose
__global__ __launch_bounds__(256) void k_pack_nchw(
    const float* __restrict__ x, int Cin, int HW,
    const float* __restrict__ mean, const float* __restrict__ rstd,
    const float* __restrict__ g, const float* __restrict__ bb,
    unsigned short* __restrict__ Anorm, unsigned short* __restrict__ Araw,
    int strideA, int c0)
{
    __shared__ float tn[32][33];
    __shared__ float tr[32][33];
    int tx = threadIdx.x & 31, ty = threadIdx.x >> 5;
    int mbase = blockIdx.x * 32, cbase = blockIdx.y * 32;
#pragma unroll
    for (int i = 0; i < 4; ++i) {
        int c = cbase + ty + 8 * i;
        int m = mbase + tx;
        int b = m / HW, pix = m - b * HW;
        float v = x[((size_t)b * Cin + c) * HW + pix];
        tn[ty + 8 * i][tx] = (v - mean[c]) * rstd[c] * g[c] + bb[c];
        tr[ty + 8 * i][tx] = v;
    }
    __syncthreads();
#pragma unroll
    for (int i = 0; i < 4; ++i) {
        int m = mbase + ty + 8 * i;
        int c = cbase + tx;
        Anorm[(size_t)m * strideA + c0 + c] = f2bf(tn[tx][ty + 8 * i]);
        if (Araw) Araw[(size_t)m * strideA + c0 + c] = f2bf(tr[tx][ty + 8 * i]);
    }
}

// weights (N x K) float -> (K x N) bf16
__global__ void k_weight_T(const float* __restrict__ w, unsigned short* __restrict__ Bm,
                           int N, int K)
{
    int id = blockIdx.x * 256 + threadIdx.x;
    if (id >= N * K) return;
    int k = id / N, n = id - k * N;
    Bm[id] = f2bf(w[(size_t)n * K + k]);
}

// per-column stats of row-major M x N float
__global__ void k_col_stats(const float* __restrict__ Y, int M, int N,
                            float* __restrict__ mean, float* __restrict__ rstd)
{
    int n = blockIdx.x * 32 + (threadIdx.x & 31);
    int ty = threadIdx.x >> 5;
    float s = 0.f, s2 = 0.f;
    for (int m = ty; m < M; m += 8) { float v = Y[(size_t)m * N + n]; s += v; s2 += v * v; }
    __shared__ float s1[8][32], q2[8][32];
    s1[ty][threadIdx.x & 31] = s; q2[ty][threadIdx.x & 31] = s2;
    __syncthreads();
    if (ty == 0) {
        float S = 0, S2 = 0;
#pragma unroll
        for (int r = 0; r < 8; ++r) { S += s1[r][threadIdx.x & 31]; S2 += q2[r][threadIdx.x & 31]; }
        float m = S / M, var = S2 / M - m * m;
        mean[n] = m; rstd[n] = rsqrtf(var + 1e-5f);
    }
}

// BN affine + swish, float row-major -> bf16 row-major
__global__ void k_bn_swish_pack(const float* __restrict__ Y, int N,
    const float* __restrict__ mean, const float* __restrict__ rstd,
    const float* __restrict__ g, const float* __restrict__ bb,
    unsigned short* __restrict__ Out, int total)
{
    int id = blockIdx.x * 256 + threadIdx.x;
    if (id >= total) return;
    int n = id % N;
    float v = (Y[id] - mean[n]) * rstd[n] * g[n] + bb[n];
    v = v / (1.f + __expf(-v));
    Out[id] = f2bf(v);
}

// depthwise 5x5, pad 2, NHWC bf16 input, float output
__global__ void k_dwconv5(const unsigned short* __restrict__ Hin,
                          const float* __restrict__ wd, const float* __restrict__ bd,
                          float* __restrict__ Yout, int E)
{
    int id = blockIdx.x * 256 + threadIdx.x;
    if (id >= MM_ * E) return;
    int e = id % E, m = id / E;
    int pix = m & (HW_ - 1);
    int y = pix >> 6, x = pix & 63;
    float acc = bd[e];
#pragma unroll
    for (int dy = -2; dy <= 2; ++dy) {
        int yy = y + dy; if (yy < 0 || yy >= HH_) continue;
#pragma unroll
        for (int dx = -2; dx <= 2; ++dx) {
            int xx = x + dx; if (xx < 0 || xx >= WW_) continue;
            acc += bf2f(Hin[(size_t)(m + dy * WW_ + dx) * E + e]) *
                   wd[e * 25 + (dy + 2) * 5 + (dx + 2)];
        }
    }
    Yout[(size_t)m * E + e] = acc;
}

// per-(batch,channel) spatial mean of row-major [M][C] float
__global__ void k_bcmean_f32(const float* __restrict__ Y, int C, float* __restrict__ pooled)
{
    int b = blockIdx.x;
    int c = blockIdx.y * 32 + (threadIdx.x & 31);
    int ty = threadIdx.x >> 5;
    float s = 0.f;
    for (int i = ty; i < HW_; i += 8) s += Y[((size_t)(b * HW_ + i)) * C + c];
    __shared__ float sh[8][32];
    sh[ty][threadIdx.x & 31] = s;
    __syncthreads();
    if (ty == 0) {
        float t = 0;
#pragma unroll
        for (int r = 0; r < 8; ++r) t += sh[r][threadIdx.x & 31];
        pooled[b * C + c] = t * (1.f / HW_);
    }
}

__global__ void k_bcmean_bf16(const unsigned short* __restrict__ X, int C, float* __restrict__ pooled)
{
    int b = blockIdx.x;
    int c = blockIdx.y * 32 + (threadIdx.x & 31);
    int ty = threadIdx.x >> 5;
    float s = 0.f;
    for (int i = ty; i < HW_; i += 8) s += bf2f(X[((size_t)(b * HW_ + i)) * C + c]);
    __shared__ float sh[8][32];
    sh[ty][threadIdx.x & 31] = s;
    __syncthreads();
    if (ty == 0) {
        float t = 0;
#pragma unroll
        for (int r = 0; r < 8; ++r) t += sh[r][threadIdx.x & 31];
        pooled[b * C + c] = t * (1.f / HW_);
    }
}

// Squeeze-excite: pooled(bn3-affined) -> relu fc1 -> sigmoid fc2
__global__ void k_se(const float* __restrict__ pooled,
                     const float* __restrict__ mean3, const float* __restrict__ rstd3,
                     const float* __restrict__ g3, const float* __restrict__ b3,
                     const float* __restrict__ w1, const float* __restrict__ b1,
                     const float* __restrict__ w2, const float* __restrict__ b2,
                     float* __restrict__ svec, int C, int hid)
{
    int b = blockIdx.x, t = threadIdx.x;
    __shared__ float pb[256];
    __shared__ float sh[64];
    if (t < C) pb[t] = (pooled[b * C + t] - mean3[t]) * rstd3[t] * g3[t] + b3[t];
    __syncthreads();
    if (t < hid) {
        float a = b1[t];
        for (int c = 0; c < C; ++c) a += pb[c] * w1[t * C + c];
        sh[t] = fmaxf(a, 0.f);
    }
    __syncthreads();
    if (t < C) {
        float a = b2[t];
        for (int h = 0; h < hid; ++h) a += sh[h] * w2[t * hid + h];
        svec[b * C + t] = 1.f / (1.f + __expf(-a));
    }
}

// x_raw + bn3(Y3) * s  -> bf16 NHWC
__global__ void k_residual(const unsigned short* __restrict__ Xraw, const float* __restrict__ Y3,
    const float* __restrict__ mean, const float* __restrict__ rstd,
    const float* __restrict__ g, const float* __restrict__ bb,
    const float* __restrict__ svec, unsigned short* __restrict__ Xout, int C, int total)
{
    int id = blockIdx.x * 256 + threadIdx.x;
    if (id >= total) return;
    int c = id % C, m = id / C, b = m >> 12;
    float h = (Y3[id] - mean[c]) * rstd[c] * g[c] + bb[c];
    float v = bf2f(Xraw[id]) + h * svec[b * C + c];
    Xout[id] = f2bf(v);
}

// latent 1x1 conv: z[m][lc] (== flat layout b*32768 + pix*8 + lc)
__global__ void k_latconv(const unsigned short* __restrict__ Xq, const float* __restrict__ lw,
                          const float* __restrict__ lb, float* __restrict__ z)
{
    int lc = threadIdx.x & 7, mi = threadIdx.x >> 3;
    int m = blockIdx.x * 32 + mi;
    float a = lb[lc];
    for (int k = 0; k < INC_; ++k)
        a += bf2f(Xq[(size_t)m * INC_ + k]) * lw[lc * INC_ + k];
    z[(size_t)m * LC_ + lc] = a;
}

__global__ void k_f2(const float* __restrict__ z, float* __restrict__ f2)
{
    int b = blockIdx.x;
    float s = 0.f;
    for (int i = threadIdx.x; i < DD_; i += 256) { float v = z[(size_t)b * DD_ + i]; s += v * v; }
    __shared__ float sd[256];
    sd[threadIdx.x] = s; __syncthreads();
    for (int st = 128; st > 0; st >>= 1) {
        if (threadIdx.x < st) sd[threadIdx.x] += sd[threadIdx.x + st];
        __syncthreads();
    }
    if (threadIdx.x == 0) f2[b] = sd[0];
}

// VQ distance: dist[b][k] = |f_b|^2 - 2 f_b.E_k + |E_k|^2  (one block per k, streams E once)
__global__ __launch_bounds__(256) void k_vqdist(const float* __restrict__ z,
                                                const float* __restrict__ Emb,
                                                const float* __restrict__ f2,
                                                float* __restrict__ dist)
{
    int k = blockIdx.x, tid = threadIdx.x;
    float acc[8] = {0, 0, 0, 0, 0, 0, 0, 0};
    float e2 = 0.f;
    const float* ek = Emb + (size_t)k * DD_;
    for (int i = tid; i < DD_; i += 256) {
        float e = ek[i];
        e2 += e * e;
#pragma unroll
        for (int b = 0; b < 8; ++b) acc[b] += z[(size_t)b * DD_ + i] * e;
    }
    __shared__ float sd[256];
    __shared__ float tmp[9];
    for (int j = 0; j < 9; ++j) {
        sd[tid] = (j < 8) ? acc[j] : e2;
        __syncthreads();
        for (int st = 128; st > 0; st >>= 1) {
            if (tid < st) sd[tid] += sd[tid + st];
            __syncthreads();
        }
        if (tid == 0) tmp[j] = sd[0];
        __syncthreads();
    }
    if (tid == 0) {
        float ek2 = tmp[8];
#pragma unroll
        for (int b = 0; b < 8; ++b) dist[b * KK_ + k] = f2[b] - 2.f * tmp[b] + ek2;
    }
}

__global__ __launch_bounds__(512) void k_argmin(const float* __restrict__ dist, int* __restrict__ idx)
{
    int t = threadIdx.x;
    __shared__ float sv[512];
    __shared__ int   si[512];
    for (int b = 0; b < 8; ++b) {
        sv[t] = dist[b * KK_ + t]; si[t] = t;
        __syncthreads();
        for (int st = 256; st > 0; st >>= 1) {
            if (t < st && sv[t + st] < sv[t]) { sv[t] = sv[t + st]; si[t] = si[t + st]; }
            __syncthreads();
        }
        if (t == 0) idx[b] = si[0];
        __syncthreads();
    }
}

__global__ void k_vqloss(const float* __restrict__ z, const float* __restrict__ Emb,
                         const int* __restrict__ idx, float* __restrict__ accum)
{
    int i = blockIdx.x * 256 + threadIdx.x;   // < 262144
    int b = i >> 15, d = i & (DD_ - 1);
    float diff = Emb[(size_t)idx[b] * DD_ + d] - z[i];
    float v = diff * diff;
    __shared__ float sd[256];
    sd[threadIdx.x] = v; __syncthreads();
    for (int st = 128; st > 0; st >>= 1) {
        if (threadIdx.x < st) sd[threadIdx.x] += sd[threadIdx.x + st];
        __syncthreads();
    }
    if (threadIdx.x == 0) atomicAdd(accum, sd[0]);
}

__global__ __launch_bounds__(512) void k_perplex(const int* __restrict__ idx, float* __restrict__ perp2)
{
    int k = threadIdx.x;
    int cnt = 0;
#pragma unroll
    for (int b = 0; b < 8; ++b) cnt += (idx[b] == k) ? 1 : 0;
    float p = cnt * 0.125f;
    __shared__ float s1[512], s2[512];
    s1[k] = p * logf(p + 1e-10f);
    s2[k] = (cnt > 0) ? 1.f : 0.f;
    __syncthreads();
    for (int st = 256; st > 0; st >>= 1) {
        if (k < st) { s1[k] += s1[k + st]; s2[k] += s2[k + st]; }
        __syncthreads();
    }
    if (k == 0) { perp2[0] = __expf(-s1[0]); perp2[1] = s2[0] / (float)KK_; }
}

__global__ __launch_bounds__(512) void k_logits_sample(const float* __restrict__ pooled,
    const float* __restrict__ logw, const float* __restrict__ logb,
    const int* __restrict__ idx, float* __restrict__ samp)
{
    int b = blockIdx.x, k = threadIdx.x;
    __shared__ float pb[128];
    __shared__ float red[512];
    __shared__ float mx, sm;
    if (k < 128) pb[k] = pooled[b * 128 + k];
    __syncthreads();
    float lg = logb[k];
    for (int c = 0; c < 128; ++c) lg += pb[c] * logw[k * 128 + c];
    red[k] = lg; __syncthreads();
    for (int st = 256; st > 0; st >>= 1) {
        if (k < st) red[k] = fmaxf(red[k], red[k + st]);
        __syncthreads();
    }
    if (k == 0) mx = red[0];
    __syncthreads();
    red[k] = __expf(lg - mx); __syncthreads();
    for (int st = 256; st > 0; st >>= 1) {
        if (k < st) red[k] += red[k + st];
        __syncthreads();
    }
    if (k == 0) sm = red[0];
    __syncthreads();
    if (k == idx[b]) samp[b] = -(lg - mx - logf(sm));
}

// concat([quantized(=E[idx]), previous]) as NHWC bf16 [M][136]
__global__ void k_ccat(const unsigned short* __restrict__ rawprev, const float* __restrict__ Emb,
                       const int* __restrict__ idx, unsigned short* __restrict__ ccat, int total)
{
    int id = blockIdx.x * 256 + threadIdx.x;
    if (id >= total) return;
    int ic = id % 136, m = id / 136;
    int b = m >> 12, pix = m & (HW_ - 1);
    if (ic < LC_) ccat[id] = f2bf(Emb[(size_t)idx[b] * DD_ + pix * LC_ + ic]);
    else          ccat[id] = rawprev[(size_t)m * PC_ + (ic - LC_)];
}

// im2col for one stride-parity phase of the transposed conv: A[m][k], K=544
__global__ void k_im2col(const unsigned short* __restrict__ ccat, unsigned short* __restrict__ Aim,
                         int py, int px, int total)
{
    int id = blockIdx.x * 256 + threadIdx.x;
    if (id >= total) return;
    int k = id % 544, m = id / 544;
    int grp = k / 136, ic = k - grp * 136;
    int dy = grp >> 1, dx = grp & 1;
    int b = m >> 12, j = m & (HW_ - 1);
    int jy = j >> 6, jx = j & 63;
    int iy = jy + py - 1 + dy;
    int ix = jx + px - 1 + dx;
    unsigned short v = 0;
    if (iy >= 0 && iy < HH_ && ix >= 0 && ix < WW_)
        v = ccat[((size_t)(b * HW_ + iy * WW_ + ix)) * 136 + ic];
    Aim[id] = v;
}

// 4 phase weight matrices: Bp[ph][k][oc] = up_w[ic][oc][3-(py+2dy)][3-(px+2dx)]
__global__ void k_upw_pack(const float* __restrict__ upw, unsigned short* __restrict__ Bp, int total)
{
    int id = blockIdx.x * 256 + threadIdx.x;
    if (id >= total) return;
    int ph = id / (544 * 128);
    int rem = id - ph * 544 * 128;
    int k = rem / 128, oc = rem - k * 128;
    int grp = k / 136, ic = k - grp * 136;
    int dy = grp >> 1, dx = grp & 1;
    int py = ph >> 1, px = ph & 1;
    int ky = py + 2 * dy, kx = px + 2 * dx;
    Bp[id] = f2bf(upw[(((size_t)ic * 128 + oc) * 4 + (3 - ky)) * 4 + (3 - kx)]);
}

// scatter phase GEMM output Yt[m][oc] into NCHW out, LDS transpose for coalescing
__global__ __launch_bounds__(256) void k_scatter_tconv(const float* __restrict__ Yt,
                                                       float* __restrict__ out, int py, int px)
{
    __shared__ float tile[32][33];
    int tx = threadIdx.x & 31, ty = threadIdx.x >> 5;
    int mbase = blockIdx.x * 32, ocbase = blockIdx.y * 32;
#pragma unroll
    for (int i = 0; i < 4; ++i) {
        int m = mbase + ty + 8 * i;
        tile[ty + 8 * i][tx] = Yt[(size_t)m * 128 + ocbase + tx];
    }
    __syncthreads();
#pragma unroll
    for (int i = 0; i < 4; ++i) {
        int m = mbase + tx;
        int oc = ocbase + ty + 8 * i;
        int b = m >> 12, j = m & (HW_ - 1);
        int jy = j >> 6, jx = j & 63;
        int oy = 2 * jy + py, ox = 2 * jx + px;
        out[(((size_t)(b * 128 + oc)) * 128 + oy) * 128 + ox] = tile[tx][ty + 8 * i];
    }
}

__global__ void k_finalize(const float* __restrict__ sqsum, const float* __restrict__ samp,
                           const float* __restrict__ perp2, float* __restrict__ outs)
{
    float msq = sqsum[0] / (float)(BN_ * DD_);
    outs[0] = msq + 0.25f * msq;        // q_loss + 0.25*e_loss (numerically equal)
    float s = 0.f;
    for (int b = 0; b < 8; ++b) s += samp[b];
    outs[1] = s * 0.125f;
    outs[2] = perp2[0];
    outs[3] = perp2[1];
}

// ---------------------------------------------------------------------------
// Host orchestration
// ---------------------------------------------------------------------------
struct CellParams {
    const float *bn0_g, *bn0_b, *w1, *b1, *bn1_g, *bn1_b, *wd, *bd,
                *bn2_g, *bn2_b, *w2, *b2, *bn3_g, *bn3_b,
                *se_w1, *se_b1, *se_w2, *se_b2;
};

static void run_cell(hipStream_t stream, const CellParams& p,
                     const float* x0, const float* x1, int C, int E, int hid,
                     unsigned short* bufA, unsigned short* bufB, float* bufY, unsigned short* bufH,
                     unsigned short* Xraw, unsigned short* Xout,
                     float* stats, float* pooled, float* svec)
{
    const int M = MM_;
    float* mean0 = stats + 0 * 768; float* rstd0 = stats + 1 * 768;
    float* mean1 = stats + 2 * 768; float* rstd1 = stats + 3 * 768;
    float* mean2 = stats + 4 * 768; float* rstd2 = stats + 5 * 768;
    float* mean3 = stats + 6 * 768; float* rstd3 = stats + 7 * 768;

    int C0 = x1 ? C / 2 : C;
    k_channel_stats<<<C0, 256, 0, stream>>>(x0, C0, HW_, BN_, mean0, rstd0);
    if (x1) k_channel_stats<<<C0, 256, 0, stream>>>(x1, C0, HW_, BN_, mean0 + C0, rstd0 + C0);

    dim3 pg(M / 32, C0 / 32);
    k_pack_nchw<<<pg, 256, 0, stream>>>(x0, C0, HW_, mean0, rstd0, p.bn0_g, p.bn0_b,
                                        bufA, Xraw, C, 0);
    if (x1)
        k_pack_nchw<<<pg, 256, 0, stream>>>(x1, C0, HW_, mean0 + C0, rstd0 + C0,
                                            p.bn0_g + C0, p.bn0_b + C0, bufA, Xraw, C, C0);

    // conv1 (expand): Y1 = A(M x C) * W1^T(C x E) + b1
    k_weight_T<<<(E * C + 255) / 256, 256, 0, stream>>>(p.w1, bufB, E, C);
    k_gemm<<<dim3(E / 128, M / 128), 256, 0, stream>>>(bufA, bufB, p.b1, bufY, M, E, C);
    k_col_stats<<<E / 32, 256, 0, stream>>>(bufY, M, E, mean1, rstd1);
    k_bn_swish_pack<<<(M * E + 255) / 256, 256, 0, stream>>>(bufY, E, mean1, rstd1,
                                                             p.bn1_g, p.bn1_b, bufH, M * E);
    // depthwise 5x5
    k_dwconv5<<<(M * E + 255) / 256, 256, 0, stream>>>(bufH, p.wd, p.bd, bufY, E);
    k_col_stats<<<E / 32, 256, 0, stream>>>(bufY, M, E, mean2, rstd2);
    k_bn_swish_pack<<<(M * E + 255) / 256, 256, 0, stream>>>(bufY, E, mean2, rstd2,
                                                             p.bn2_g, p.bn2_b, bufA, M * E);
    // conv2 (project): Y3 = A2(M x E) * W2^T(E x C) + b2
    k_weight_T<<<(C * E + 255) / 256, 256, 0, stream>>>(p.w2, bufB, C, E);
    k_gemm<<<dim3(C / 128, M / 128), 256, 0, stream>>>(bufA, bufB, p.b2, bufY, M, C, E);
    k_col_stats<<<C / 32, 256, 0, stream>>>(bufY, M, C, mean3, rstd3);

    // SE (mean commutes with affine bn3)
    k_bcmean_f32<<<dim3(BN_, C / 32), 256, 0, stream>>>(bufY, C, pooled);
    k_se<<<BN_, 256, 0, stream>>>(pooled, mean3, rstd3, p.bn3_g, p.bn3_b,
                                  p.se_w1, p.se_b1, p.se_w2, p.se_b2, svec, C, hid);
    k_residual<<<(M * C + 255) / 256, 256, 0, stream>>>(Xraw, bufY, mean3, rstd3,
                                                        p.bn3_g, p.bn3_b, svec, Xout, C, M * C);
}

extern "C" void kernel_launch(void* const* d_in, const int* in_sizes, int n_in,
                              void* d_out, int out_size, void* d_ws, size_t ws_size,
                              hipStream_t stream)
{
    (void)in_sizes; (void)n_in; (void)out_size; (void)ws_size;
    auto F = [&](int k) { return (const float*)d_in[k]; };

    const float* previous = F(0);
    const float* feature  = F(1);
    CellParams cq = { F(2), F(3), F(4), F(5), F(6), F(7), F(8), F(9), F(10), F(11),
                      F(12), F(13), F(14), F(15), F(16), F(17), F(18), F(19) };
    const float* lat_w = F(20); const float* lat_b = F(21);
    const float* emb   = F(22);
    const float* up_w  = F(23); const float* up_b  = F(24);
    CellParams cl = { F(25), F(26), F(27), F(28), F(29), F(30), F(31), F(32), F(33), F(34),
                      F(35), F(36), F(37), F(38), F(39), F(40), F(41), F(42) };
    const float* log_w = F(43); const float* log_b = F(44);

    // ---- workspace carve-up ----
    char* ws = (char*)d_ws;
    auto alloc = [&](size_t bytes) { char* p = ws; ws += (bytes + 255) & ~(size_t)255; return p; };

    unsigned short* bufA   = (unsigned short*)alloc((size_t)MM_ * 768 * 2);   // 50.3 MB
    unsigned short* bufH   = (unsigned short*)alloc((size_t)MM_ * 768 * 2);   // 50.3 MB
    float*          bufY   = (float*)alloc((size_t)MM_ * 768 * 4);            // 100.7 MB
    unsigned short* XrawQ  = (unsigned short*)alloc((size_t)MM_ * INC_ * 2);  // 16.8 MB
    unsigned short* Xq     = (unsigned short*)alloc((size_t)MM_ * INC_ * 2);  // 16.8 MB
    unsigned short* XrawL  = (unsigned short*)alloc((size_t)MM_ * PC_ * 2);   // 8.4 MB
    unsigned short* Xl     = (unsigned short*)alloc((size_t)MM_ * PC_ * 2);   // 8.4 MB
    unsigned short* bufB   = (unsigned short*)alloc(2u * 1024 * 1024);        // weight mats
    unsigned short* ccat   = (unsigned short*)alloc((size_t)MM_ * 136 * 2);   // 8.9 MB
    float*          zbuf   = (float*)alloc((size_t)BN_ * DD_ * 4);            // 1 MB
    float*          stats  = (float*)alloc(8 * 768 * 4);
    float*          pooled = (float*)alloc(BN_ * 256 * 4);
    float*          svec   = (float*)alloc(BN_ * 256 * 4);
    float*          distb  = (float*)alloc(BN_ * KK_ * 4);
    float*          f2b    = (float*)alloc(BN_ * 4);
    int*            idxb   = (int*)alloc(BN_ * 4);
    float*          sqsum  = (float*)alloc(4);
    float*          samp   = (float*)alloc(BN_ * 4);
    float*          perp2  = (float*)alloc(2 * 4);

    float* out_t   = (float*)d_out;
    float* out_scl = out_t + (size_t)BN_ * 128 * 128 * 128;

    // ---- cell_q on concat(previous, feature) ----
    run_cell(stream, cq, previous, feature, INC_, 768, 16,
             bufA, bufB, bufY, bufH, XrawQ, Xq, stats, pooled, svec);

    // ---- latent + VQ ----
    k_latconv<<<MM_ / 32, 256, 0, stream>>>(Xq, lat_w, lat_b, zbuf);
    k_f2<<<BN_, 256, 0, stream>>>(zbuf, f2b);
    k_vqdist<<<KK_, 256, 0, stream>>>(zbuf, emb, f2b, distb);
    k_argmin<<<1, 512, 0, stream>>>(distb, idxb);
    hipMemsetAsync(sqsum, 0, sizeof(float), stream);
    k_vqloss<<<(BN_ * DD_) / 256, 256, 0, stream>>>(zbuf, emb, idxb, sqsum);
    k_perplex<<<1, 512, 0, stream>>>(idxb, perp2);

    // ---- cell_l on previous ----
    run_cell(stream, cl, previous, nullptr, PC_, 384, 8,
             bufA, bufB, bufY, bufH, XrawL, Xl, stats, pooled, svec);
    k_bcmean_bf16<<<dim3(BN_, PC_ / 32), 256, 0, stream>>>(Xl, PC_, pooled);
    k_logits_sample<<<BN_, 512, 0, stream>>>(pooled, log_w, log_b, idxb, samp);

    // ---- transposed conv (4 stride-parity phases as WMMA GEMMs) ----
    k_ccat<<<(MM_ * 136 + 255) / 256, 256, 0, stream>>>(XrawL, emb, idxb, ccat, MM_ * 136);
    k_upw_pack<<<(4 * 544 * 128 + 255) / 256, 256, 0, stream>>>(up_w, bufB, 4 * 544 * 128);
    for (int ph = 0; ph < 4; ++ph) {
        int py = ph >> 1, px = ph & 1;
        k_im2col<<<(MM_ * 544 + 255) / 256, 256, 0, stream>>>(ccat, bufA, py, px, MM_ * 544);
        k_gemm<<<dim3(1, MM_ / 128), 256, 0, stream>>>(bufA, bufB + (size_t)ph * 544 * 128,
                                                       up_b, bufY, MM_, 128, 544);
        k_scatter_tconv<<<dim3(MM_ / 32, 4), 256, 0, stream>>>(bufY, out_t, py, px);
    }

    // ---- scalars ----
    k_finalize<<<1, 1, 0, stream>>>(sqsum, samp, perp2, out_scl);
}
